// MultiHeadSelfAttention_10514079941419
// MI455X (gfx1250) — matrix operand order repaired and checked by reading, not compiled, not run on hardware
//
#include <hip/hip_runtime.h>
#include <cstdint>
#include <cstddef>

#define BS  4
#define SEQ 4096
#define DIM 768
#define NH  12
#define HD  64
#define LKK 256
#define WIN 16

typedef __bf16 bf16_t;
typedef __attribute__((ext_vector_type(8)))  __bf16 v8bf;
typedef __attribute__((ext_vector_type(16))) __bf16 v16bf;
typedef __attribute__((ext_vector_type(8)))  float  v8f;

__device__ __forceinline__ v16bf join8(v8bf lo, v8bf hi) {
  return __builtin_shufflevector(lo, hi, 0,1,2,3,4,5,6,7,8,9,10,11,12,13,14,15);
}
__device__ __forceinline__ v8f wmma_bf16(v16bf a, v16bf b, v8f c) {
  // D = A(16x32) * B(32x16) + C(16x16 f32)
  return __builtin_amdgcn_wmma_f32_16x16x32_bf16(false, a, false, b, (short)0, c, false, false);
}
__device__ __forceinline__ v8f vzero8() {
  v8f v = {0.f,0.f,0.f,0.f,0.f,0.f,0.f,0.f};
  return v;
}
__device__ __forceinline__ bf16_t to_bf16(float f) { return (bf16_t)f; }

// Async 16B global->LDS DMA (GLOBAL_LOAD_ASYNC_TO_LDS_B128, ASYNCcnt-tracked).
// ldsp: generic pointer to __shared__ (low 32 bits = workgroup-relative LDS
// byte address, which is what the VDST operand wants); gp: global address.
__device__ __forceinline__ void async_copy16(const void* ldsp, const void* gp) {
  uint32_t l = (uint32_t)(uintptr_t)ldsp;
  asm volatile("global_load_async_to_lds_b128 %0, %1, off"
               :: "v"(l), "v"(gp) : "memory");
}
__device__ __forceinline__ void async_wait0() {
  asm volatile("s_wait_asynccnt 0x0" ::: "memory");
}
// Wait until <=4 async ops outstanding: current double-buffer stage (4 wave
// ops) is complete while the next stage's 4 ops continue in flight.
__device__ __forceinline__ void async_wait4() {
  asm volatile("s_wait_asynccnt 0x4" ::: "memory");
}

// ---------------------------------------------------------------------------
// Stage 1: fp32 -> bf16 conversion (query + weights)
// ---------------------------------------------------------------------------
__global__ void cvt_f32_to_bf16(const float* __restrict__ src,
                                bf16_t* __restrict__ dst, int n) {
  int i = (blockIdx.x * blockDim.x + threadIdx.x) * 4;
  if (i < n) {
    float4 f = *(const float4*)(src + i);
    dst[i + 0] = to_bf16(f.x);
    dst[i + 1] = to_bf16(f.y);
    dst[i + 2] = to_bf16(f.z);
    dst[i + 3] = to_bf16(f.w);
  }
}

// ---------------------------------------------------------------------------
// Stage 2: masked adaptive-avg-pool of raw key/value (pool commutes with the
// linear projection): keyP[b][l][:] = (1/16) * sum_w keep * key[b][l*16+w][:]
// cnts[b*LK+l] = (#kept)/16  (scales the bias in the K/V projection epilogue)
// ---------------------------------------------------------------------------
__global__ void pool_kv(const float* __restrict__ key,
                        const float* __restrict__ value,
                        const float* __restrict__ mask,
                        bf16_t* __restrict__ keyP,
                        bf16_t* __restrict__ valP,
                        float* __restrict__ cnts) {
  int bl = blockIdx.x;           // b*LK + l
  int b  = bl >> 8, l = bl & 255;
  int tid = threadIdx.x;
  __shared__ float keep[WIN];
  if (tid < WIN) keep[tid] = (mask[(size_t)b * SEQ + l * WIN + tid] >= 0.f) ? 1.f : 0.f;
  __syncthreads();
  if (tid == 0) {
    float c = 0.f;
    for (int w = 0; w < WIN; w++) c += keep[w];
    cnts[bl] = c * (1.f / WIN);
  }
  for (int c = tid; c < DIM; c += blockDim.x) {
    float sk = 0.f, sv = 0.f;
    for (int w = 0; w < WIN; w++) {
      size_t idx = ((size_t)b * SEQ + (size_t)l * WIN + w) * DIM + c;
      sk += keep[w] * key[idx];
      sv += keep[w] * value[idx];
    }
    keyP[(size_t)bl * DIM + c] = to_bf16(sk * (1.f / WIN));
    valP[(size_t)bl * DIM + c] = to_bf16(sv * (1.f / WIN));
  }
}

// ---------------------------------------------------------------------------
// Stage 3/5: bf16 WMMA GEMM, C[M][768] = A[M][768] @ W.T, K=N=768.
// Block = 128x128 tile, 256 threads = 8 waves, each wave 32x64 (2x4 WMMA tiles).
// W is row-major [N][K] == exactly the B^T layout the B-fragment needs.
// Double-buffered async DMA staging: stage k+1 while WMMAs consume k
// (s_wait_asynccnt 4 drains only the current buffer's 4 wave-ops).
// Epilogues: 0=Q-proj (scale 1/8, head-split), 1=K-proj (cnt-scaled bias,
// [bh][256][64]), 2=V-proj ([bh][64][256] transposed), 3=out-proj (fp32).
// ---------------------------------------------------------------------------
#define LDSTR 40   // padded LDS row stride (bf16 elems) -> conflict-free frags
#define KSTEPS (DIM / 32)

template <int EPI>
__global__ void gemm_bf16_k768(const bf16_t* __restrict__ A,
                               const bf16_t* __restrict__ W,
                               const float* __restrict__ bias,
                               const float* __restrict__ cnts,
                               void* __restrict__ outp) {
  __shared__ bf16_t ldsA[2][128 * LDSTR];
  __shared__ bf16_t ldsB[2][128 * LDSTR];
  const int tid  = threadIdx.x;
  const int lane = tid & 31, wave = tid >> 5;
  const int lrow = lane & 15, lhi = lane >> 4;
  const int n0 = blockIdx.x * 128;
  const int m0 = blockIdx.y * 128;
  const int wrb = (wave & 3) * 32;   // wave row base in tile
  const int wcb = (wave >> 2) * 64;  // wave col base in tile

  v8f acc[2][4];
  for (int i = 0; i < 2; i++)
    for (int j = 0; j < 4; j++) acc[i][j] = vzero8();

  // async-stage 128x32 of A and of B^T(=W rows) into buffer `buf`:
  // 2 chunks/thread x 2 matrices = 4 async wave-ops per wave per stage
  auto stage = [&](int buf, int k0) {
    for (int c = tid; c < 512; c += 256) {
      int row = c >> 2, kc = (c & 3) * 8;
      async_copy16(&ldsA[buf][row * LDSTR + kc],
                   &A[(size_t)(m0 + row) * DIM + k0 + kc]);
      async_copy16(&ldsB[buf][row * LDSTR + kc],
                   &W[(size_t)(n0 + row) * DIM + k0 + kc]);
      if (k0 + 64 < DIM) {  // L2 prefetch 2 k-steps ahead -> global_prefetch_b8
        __builtin_prefetch(&A[(size_t)(m0 + row) * DIM + k0 + 64 + kc], 0, 0);
        __builtin_prefetch(&W[(size_t)(n0 + row) * DIM + k0 + 64 + kc], 0, 0);
      }
    }
  };

  stage(0, 0);
  for (int ks = 0; ks < KSTEPS; ks++) {
    const int buf = ks & 1;
    if (ks + 1 < KSTEPS) {
      stage(buf ^ 1, (ks + 1) * 32);  // overlap next DMA with this compute
      async_wait4();                  // current buffer complete, next in flight
    } else {
      async_wait0();
    }
    __syncthreads();
    v16bf afr[2], bfr[4];
    for (int mt = 0; mt < 2; mt++) {
      const bf16_t* ap = &ldsA[buf][(wrb + mt * 16 + lrow) * LDSTR + lhi * 16];
      afr[mt] = join8(*(const v8bf*)ap, *(const v8bf*)(ap + 8));
    }
    for (int nt = 0; nt < 4; nt++) {
      const bf16_t* bp = &ldsB[buf][(wcb + nt * 16 + lrow) * LDSTR + lhi * 16];
      bfr[nt] = join8(*(const v8bf*)bp, *(const v8bf*)(bp + 8));
    }
    for (int mt = 0; mt < 2; mt++)
      for (int nt = 0; nt < 4; nt++)
        acc[mt][nt] = wmma_bf16(afr[mt], bfr[nt], acc[mt][nt]);
    __syncthreads();  // all waves done reading `buf` before it is re-staged
  }

  for (int mt = 0; mt < 2; mt++)
    for (int nt = 0; nt < 4; nt++)
      for (int r = 0; r < 8; r++) {
        int m = m0 + wrb + mt * 16 + lhi * 8 + r;  // C layout: row = r + 8*(lane/16)
        int n = n0 + wcb + nt * 16 + lrow;         //           col = lane%16
        float v = acc[mt][nt][r];
        if constexpr (EPI == 0) {
          v = (v + bias[n]) * 0.125f;  // 1/sqrt(64) fused
          int b = m >> 12, q = m & 4095;
          int h = n >> 6, d = n & 63;
          ((bf16_t*)outp)[(((size_t)b * NH + h) * SEQ + q) * HD + d] = to_bf16(v);
        } else if constexpr (EPI == 1) {
          v = v + bias[n] * cnts[m];
          int b = m >> 8, kx = m & 255;
          int h = n >> 6, d = n & 63;
          ((bf16_t*)outp)[(((size_t)b * NH + h) * LKK + kx) * HD + d] = to_bf16(v);
        } else if constexpr (EPI == 2) {
          v = v + bias[n] * cnts[m];
          int b = m >> 8, kx = m & 255;
          int h = n >> 6, d = n & 63;
          ((bf16_t*)outp)[(((size_t)b * NH + h) * HD + d) * LKK + kx] = to_bf16(v);
        } else {
          ((float*)outp)[(size_t)m * DIM + n] = v + bias[n];
        }
      }
}

// ---------------------------------------------------------------------------
// Stage 4: fused low-rank attention per (b,h), flash-style online softmax over
// 8 key tiles of 32 so live register state stays ~100 VGPRs (no scratch
// spills). Q/K/V^T staged by async DMA; P tile bounces through wave-private
// LDS (same-wave DS ops are in-order -> no barrier) to convert accumulator
// layout -> A-fragment layout.
// ---------------------------------------------------------------------------
#define PADQ 72
#define PADK 72
#define PADV 264
#define PADW 40

__global__ void attn_kernel(const bf16_t* __restrict__ Qh,
                            const bf16_t* __restrict__ Kc,
                            const bf16_t* __restrict__ Vt,
                            bf16_t* __restrict__ ctx) {
  extern __shared__ char smem[];
  bf16_t* sQ  = (bf16_t*)smem;          // 128 x PADQ
  bf16_t* sK  = sQ + 128 * PADQ;        // 256 x PADK   (B^T for S = Q@K^T)
  bf16_t* sV  = sK + 256 * PADK;        // 64  x PADV   (V^T = B^T for P@V)
  bf16_t* sPw = sV + 64 * PADV;         // 8 waves x (16 x PADW) P staging

  const int bh = blockIdx.x;           // b*NH + h
  const int q0 = blockIdx.y * 128;
  const int tid = threadIdx.x, lane = tid & 31, wave = tid >> 5;
  const int lrow = lane & 15, lhi = lane >> 4;
  const int mrow = wave * 16;          // each wave owns 16 query rows
  bf16_t* myP = sPw + wave * 16 * PADW;

  const bf16_t* Qb = Qh + ((size_t)bh * SEQ + q0) * HD;
  const bf16_t* Kb = Kc + (size_t)bh * LKK * HD;
  const bf16_t* Vb = Vt + (size_t)bh * HD * LKK;

  for (int c = tid; c < 128 * 8; c += 256) {   // Q tile 128x64
    int row = c >> 3, kc = (c & 7) * 8;
    async_copy16(&sQ[row * PADQ + kc], &Qb[(size_t)row * HD + kc]);
  }
  for (int c = tid; c < 256 * 8; c += 256) {   // K 256x64
    int row = c >> 3, kc = (c & 7) * 8;
    async_copy16(&sK[row * PADK + kc], &Kb[(size_t)row * HD + kc]);
  }
  for (int c = tid; c < 64 * 32; c += 256) {   // V^T 64x256
    int row = c >> 5, kc = (c & 31) * 8;
    async_copy16(&sV[row * PADV + kc], &Vb[(size_t)row * LKK + kc]);
  }
  async_wait0();
  __syncthreads();

  v8f cacc[4];
  for (int nt = 0; nt < 4; nt++) cacc[nt] = vzero8();
  float rmax[8], rsum[8];
  for (int r = 0; r < 8; r++) { rmax[r] = -3.0e38f; rsum[r] = 0.f; }

  for (int kt = 0; kt < LKK / 32; kt++) {      // 8 tiles of 32 keys
    // ---- scores for 16 rows x 32 keys (K-dim = 64 in 2 wmma steps) ----
    v8f s0 = vzero8(), s1 = vzero8();
    for (int k0 = 0; k0 < HD; k0 += 32) {
      const bf16_t* ap = &sQ[(mrow + lrow) * PADQ + k0 + lhi * 16];
      v16bf afr = join8(*(const v8bf*)ap, *(const v8bf*)(ap + 8));
      const bf16_t* bp0 = &sK[(kt * 32 + lrow) * PADK + k0 + lhi * 16];
      const bf16_t* bp1 = &sK[(kt * 32 + 16 + lrow) * PADK + k0 + lhi * 16];
      s0 = wmma_bf16(afr, join8(*(const v8bf*)bp0, *(const v8bf*)(bp0 + 8)), s0);
      s1 = wmma_bf16(afr, join8(*(const v8bf*)bp1, *(const v8bf*)(bp1 + 8)), s1);
    }
    // ---- online softmax update; row m = mrow + 8*(lane/16) + r lives in
    //      one 16-lane half, so 4 shfl_xor steps reduce over its 16 cols ----
    for (int r = 0; r < 8; r++) {
      float mx = fmaxf(s0[r], s1[r]);
      for (int off = 8; off >= 1; off >>= 1) mx = fmaxf(mx, __shfl_xor(mx, off, 32));
      float nm = fmaxf(rmax[r], mx);
      float corr = __expf(rmax[r] - nm);
      rmax[r] = nm;
      float p0 = __expf(s0[r] - nm);
      float p1 = __expf(s1[r] - nm);
      s0[r] = p0;
      s1[r] = p1;
      float ls = p0 + p1;
      for (int off = 8; off >= 1; off >>= 1) ls += __shfl_xor(ls, off, 32);
      rsum[r] = rsum[r] * corr + ls;
      for (int nt = 0; nt < 4; nt++) cacc[nt][r] *= corr;
    }
    // ---- P tile (16x32) accumulator layout -> wave-private LDS row-major ----
    for (int r = 0; r < 8; r++) {
      myP[(lhi * 8 + r) * PADW + lrow]      = to_bf16(s0[r]);
      myP[(lhi * 8 + r) * PADW + 16 + lrow] = to_bf16(s1[r]);
    }
    // same-wave DS in-order: loads below see the stores above, no barrier
    const bf16_t* pp = &myP[lrow * PADW + lhi * 16];
    v16bf pfr = join8(*(const v8bf*)pp, *(const v8bf*)(pp + 8));
    for (int nt = 0; nt < 4; nt++) {
      const bf16_t* bp = &sV[(nt * 16 + lrow) * PADV + kt * 32 + lhi * 16];
      v16bf bfr = join8(*(const v8bf*)bp, *(const v8bf*)(bp + 8));
      cacc[nt] = wmma_bf16(pfr, bfr, cacc[nt]);
    }
  }

  const int b = bh / NH, h = bh % NH;
  for (int nt = 0; nt < 4; nt++)
    for (int r = 0; r < 8; r++) {
      int q = q0 + mrow + lhi * 8 + r;
      int d = nt * 16 + lrow;
      float v = cacc[nt][r] / rsum[r];   // final softmax normalization
      ctx[((size_t)b * SEQ + q) * DIM + h * HD + d] = to_bf16(v);
    }
}

// ---------------------------------------------------------------------------
extern "C" void kernel_launch(void* const* d_in, const int* in_sizes, int n_in,
                              void* d_out, int out_size, void* d_ws, size_t ws_size,
                              hipStream_t stream) {
  (void)in_sizes; (void)n_in; (void)out_size; (void)ws_size;
  const float* query = (const float*)d_in[0];
  const float* key   = (const float*)d_in[1];
  const float* value = (const float*)d_in[2];
  const float* mask  = (const float*)d_in[3];
  const float* Wq = (const float*)d_in[4];
  const float* bq = (const float*)d_in[5];
  const float* Wk = (const float*)d_in[6];
  const float* bk = (const float*)d_in[7];
  const float* Wv = (const float*)d_in[8];
  const float* bv = (const float*)d_in[9];
  const float* Wo = (const float*)d_in[10];
  const float* bo = (const float*)d_in[11];

  char* ws = (char*)d_ws;
  size_t off = 0;
  auto take = [&](size_t bytes) {
    char* p = ws + off;
    off = (off + bytes + 255) & ~(size_t)255;
    return p;
  };
  bf16_t* qbf  = (bf16_t*)take((size_t)BS * SEQ * DIM * 2);
  bf16_t* wqb  = (bf16_t*)take((size_t)DIM * DIM * 2);
  bf16_t* wkb  = (bf16_t*)take((size_t)DIM * DIM * 2);
  bf16_t* wvb  = (bf16_t*)take((size_t)DIM * DIM * 2);
  bf16_t* wob  = (bf16_t*)take((size_t)DIM * DIM * 2);
  bf16_t* keyP = (bf16_t*)take((size_t)BS * LKK * DIM * 2);
  bf16_t* valP = (bf16_t*)take((size_t)BS * LKK * DIM * 2);
  float*  cnts = (float*)take((size_t)BS * LKK * 4);
  bf16_t* Kc   = (bf16_t*)take((size_t)BS * NH * LKK * HD * 2);
  bf16_t* Vt   = (bf16_t*)take((size_t)BS * NH * HD * LKK * 2);
  bf16_t* Qh   = (bf16_t*)take((size_t)BS * NH * SEQ * HD * 2);
  bf16_t* ctx  = (bf16_t*)take((size_t)BS * SEQ * DIM * 2);

  const int nq = BS * SEQ * DIM;
  const int nw = DIM * DIM;
  cvt_f32_to_bf16<<<(nq / 4 + 255) / 256, 256, 0, stream>>>(query, qbf, nq);
  cvt_f32_to_bf16<<<(nw / 4 + 255) / 256, 256, 0, stream>>>(Wq, wqb, nw);
  cvt_f32_to_bf16<<<(nw / 4 + 255) / 256, 256, 0, stream>>>(Wk, wkb, nw);
  cvt_f32_to_bf16<<<(nw / 4 + 255) / 256, 256, 0, stream>>>(Wv, wvb, nw);
  cvt_f32_to_bf16<<<(nw / 4 + 255) / 256, 256, 0, stream>>>(Wo, wob, nw);

  pool_kv<<<BS * LKK, 256, 0, stream>>>(key, value, mask, keyP, valP, cnts);

  dim3 gq(DIM / 128, (BS * SEQ) / 128);   // 6 x 128
  dim3 gkv(DIM / 128, (BS * LKK) / 128);  // 6 x 8
  gemm_bf16_k768<0><<<gq, 256, 0, stream>>>(qbf, wqb, bq, nullptr, Qh);
  gemm_bf16_k768<1><<<gkv, 256, 0, stream>>>(keyP, wkb, bk, cnts, Kc);
  gemm_bf16_k768<2><<<gkv, 256, 0, stream>>>(valP, wvb, bv, cnts, Vt);

  const size_t attn_lds =
      (size_t)(128 * PADQ + 256 * PADK + 64 * PADV + 8 * 16 * PADW) * 2;
  (void)hipFuncSetAttribute((const void*)attn_kernel,
                            hipFuncAttributeMaxDynamicSharedMemorySize,
                            (int)attn_lds);
  attn_kernel<<<dim3(BS * NH, SEQ / 128), 256, attn_lds, stream>>>(Qh, Kc, Vt, ctx);

  gemm_bf16_k768<3><<<gq, 256, 0, stream>>>(ctx, wob, bo, nullptr, d_out);
}
